// VAE_11493332484853
// MI455X (gfx1250) — matrix operand order repaired and checked by reading, not compiled, not run on hardware
//
// MI455X (gfx1250) fused VAE-ELBO forward, round 3.
//
// Roofline: ~220 GFLOP, ~530 MB mandatory HBM traffic -> memory-bound floor
// ~23 us @ 23.3 TB/s. One fused kernel: each WG owns 64 rows; h / z / hd /
// logits live in LDS (~295 KB of 320 KB); weights stream from the 192 MB L2.
// GEMMs use V_WMMA_F32_16X16X32_BF16 with f32 accumulation.
//
// Round-3 changes (from asm inspection):
//  * 4x4 per-wave WMMA tiling (wave = all 4 M-blocks x one 64-col N-octant):
//    16 ds_load_b128 per wave per 32-k chunk (was 34), 2x less WG-level LDS
//    fragment traffic, and 16 back-to-back wmmas after one dscnt wait.
//  * Softmax row scans use 16B bf16x8 LDS loads.

#include <hip/hip_runtime.h>

typedef __bf16 bf16;
typedef __attribute__((ext_vector_type(16))) __bf16 v16bf;
typedef __attribute__((ext_vector_type(8)))  __bf16 v8bf;
typedef __attribute__((ext_vector_type(8)))  float  v8f;

#define DEV static __device__ __forceinline__

constexpr int   Bn     = 65536;
constexpr int   DIN    = 1024;
constexpr int   DH     = 512;
constexpr int   DZ     = 64;
constexpr int   TM     = 64;            // rows per workgroup
constexpr float LOG2PI = 1.8378770664093453f;

// LDS pitches (bf16 elements)
constexpr int PH  = 520;   // h / hd tile pitch           (64 x 512)
constexpr int PW  = 40;    // staged 32-k weight slice    ([n][k] transposed)
constexpr int PW3 = 72;    // staged 64-k weight slice    (stage 3)
constexpr int PL  = 1032;  // logits tile pitch           (64 x 1024)
constexpr int PZ  = 72;    // z tile pitch                (64 x 64)
constexpr int PX  = 40;    // x_noisy 32-k tile pitch     (64 x 32)

// LDS layout (byte offsets)
constexpr int WBUF    = 512 * PW * 2;                 // 40960 per ping-pong half
constexpr int OFF_H   = 0;                            // 66560
constexpr int OFF_W0  = OFF_H + TM * PH * 2;          // 66560
constexpr int OFF_W1  = OFF_W0 + WBUF;                // 107520
constexpr int OFF_LOG = OFF_W0 + 2 * WBUF;            // 148480
constexpr int OFF_Z   = OFF_LOG + TM * PL * 2;        // 280576
constexpr int OFF_X0  = OFF_Z + TM * PZ * 2;          // 289792
constexpr int OFF_X1  = OFF_X0 + TM * PX * 2;         // 294912
constexpr int OFF_KL  = OFF_X1 + TM * PX * 2;         // 300032
constexpr int OFF_MAX = OFF_KL + 256;
constexpr int OFF_SUM = OFF_MAX + 256;
constexpr int OFF_LP  = OFF_SUM + 256;
constexpr int OFF_RED = OFF_LP + 256;
constexpr int SMEM_BYTES = OFF_RED + 1024;            // 302080
static_assert(SMEM_BYTES <= 320 * 1024, "LDS overflow");
static_assert(512 * PW3 * 2 <= 2 * WBUF, "stage3 slice overflows W region");

DEV v8f v8f_zero() {
  v8f z;
#pragma unroll
  for (int i = 0; i < 8; ++i) z[i] = 0.f;
  return z;
}

DEV v8f wmma_bf16(v16bf a, v16bf b, v8f c) {
  return __builtin_amdgcn_wmma_f32_16x16x32_bf16(false, a, false, b,
                                                 (short)0, c, false, false);
}

// A fragment (16x32 bf16 tile, row-major in LDS).
// lane<16: K = kbase+{0..15}; lane>=16: K = kbase+{16..31} (contiguous).
DEV v16bf ldsA(const bf16* base, int row, int kbase, int lane, int pitch) {
  const bf16* p = base + row * pitch + kbase + ((lane >> 4) << 4);
  v16bf a;
#pragma unroll
  for (int i = 0; i < 16; ++i) a[i] = p[i];
  return a;
}

// B fragment (32x16 bf16 tile) from a transposed [n][k] LDS slice.
// lane L holds column N = L&15; K = kbase + 16*(L>=16) + i (contiguous).
DEV v16bf ldsB(const bf16* baseT, int nbase, int kbase, int lane, int pitch) {
  const bf16* p = baseT + (nbase + (lane & 15)) * pitch + kbase + ((lane >> 4) << 4);
  v16bf b;
#pragma unroll
  for (int i = 0; i < 16; ++i) b[i] = p[i];
  return b;
}

// Cooperative stage of W[k0+kk][n0+n] (row-major fp32) into transposed bf16
// LDS slice dst[n*pitch + kk]. Global reads are coalesced over n.
DEV void stageWT(const float* __restrict__ W, int k0, int n0, int ldw,
                 bf16* dst, int KK, int NN, int pitch, int tid) {
  for (int idx = tid; idx < KK * NN; idx += 256) {
    int kk = idx / NN;
    int n  = idx - kk * NN;
    dst[n * pitch + kk] = (bf16)W[(size_t)(k0 + kk) * ldw + n0 + n];
  }
}

// Stage a 64x32 fp32 row tile of X as bf16 (row-major, pitch PX).
DEV void stageX(const float* __restrict__ X, int row0, int k0, bf16* dst, int tid) {
  for (int idx = tid; idx < TM * 32; idx += 256) {
    int r = idx >> 5, kk = idx & 31;
    dst[r * PX + kk] = (bf16)X[(size_t)(row0 + r) * DIN + k0 + kk];
  }
}

// One 32-k chunk of the 64x512 per-WG GEMM with 4x4 per-wave tiling:
// wave w owns all four 16-row M blocks and the 64-col N-octant [w*64, w*64+64).
// 4 A-frags + 4 B-frags are loaded up front (16 ds_load_b128), then the 16
// WMMAs run back-to-back with full register reuse.
DEV void gemm_4x4(const bf16* sA, int pa, const bf16* sBt, int pb,
                  int ka, int kb, v8f acc[16], int w, int lane) {
  v16bf a[4], b[4];
#pragma unroll
  for (int mi = 0; mi < 4; ++mi)
    a[mi] = ldsA(sA, mi * 16 + (lane & 15), ka, lane, pa);
#pragma unroll
  for (int nj = 0; nj < 4; ++nj)
    b[nj] = ldsB(sBt, w * 64 + nj * 16, kb, lane, pb);
#pragma unroll
  for (int mi = 0; mi < 4; ++mi)
#pragma unroll
    for (int nj = 0; nj < 4; ++nj)
      acc[mi * 4 + nj] = wmma_bf16(a[mi], b[nj], acc[mi * 4 + nj]);
}

__global__ __launch_bounds__(256, 1)
void vae_fused(const float* __restrict__ xn,  const float* __restrict__ x,
               const float* __restrict__ eps,
               const float* __restrict__ ew1,  const float* __restrict__ eb1,
               const float* __restrict__ ewmu, const float* __restrict__ ebmu,
               const float* __restrict__ ewlv, const float* __restrict__ eblv,
               const float* __restrict__ dw1,  const float* __restrict__ db1,
               const float* __restrict__ dwmu, const float* __restrict__ dbmu,
               const float* __restrict__ dwlv, const float* __restrict__ dblv,
               float* __restrict__ partial) {
  extern __shared__ char smem[];
  bf16*  sH   = (bf16*)(smem + OFF_H);
  bf16*  sW0  = (bf16*)(smem + OFF_W0);
  bf16*  sW1  = (bf16*)(smem + OFF_W1);
  bf16*  sMu0 = (bf16*)(smem + OFF_W0);             // stage-2 sub-buffers
  bf16*  sLv0 = (bf16*)(smem + OFF_W0 + 20480);
  bf16*  sMu1 = (bf16*)(smem + OFF_W1);
  bf16*  sLv1 = (bf16*)(smem + OFF_W1 + 20480);
  bf16*  sLog = (bf16*)(smem + OFF_LOG);
  bf16*  sZ   = (bf16*)(smem + OFF_Z);
  bf16*  sX0  = (bf16*)(smem + OFF_X0);
  bf16*  sX1  = (bf16*)(smem + OFF_X1);
  float* sKL  = (float*)(smem + OFF_KL);
  float* sMax = (float*)(smem + OFF_MAX);
  float* sSum = (float*)(smem + OFF_SUM);
  float* sLP  = (float*)(smem + OFF_LP);
  float* sRed = (float*)(smem + OFF_RED);

  const int tid  = threadIdx.x;
  const int lane = tid & 31;
  const int w    = tid >> 5;
  const int row0 = blockIdx.x * TM;
  const int hi8  = (lane >> 4) << 3;   // +8 row offset for hi half-wave

  if (tid < TM) { sKL[tid] = 0.f; sLP[tid] = 0.f; }

  v8f acc[16];

  // ---------------- stage 1: h = tanh(x_noisy @ enc_w1 + b1) --------------
#pragma unroll
  for (int t = 0; t < 16; ++t) acc[t] = v8f_zero();
  stageX(xn, row0, 0, sX0, tid);
  stageWT(ew1, 0, 0, DH, sW0, 32, DH, PW, tid);
  __syncthreads();
  for (int c = 0; c < DIN / 32; ++c) {
    const bf16* Wc = (c & 1) ? sW1 : sW0;
    const bf16* Xc = (c & 1) ? sX1 : sX0;
    if (c + 1 < DIN / 32) {
      stageX(xn, row0, (c + 1) * 32, (c & 1) ? sX0 : sX1, tid);
      stageWT(ew1, (c + 1) * 32, 0, DH, (c & 1) ? sW0 : sW1, 32, DH, PW, tid);
    }
    gemm_4x4(Xc, PX, Wc, PW, 0, 0, acc, w, lane);
    __syncthreads();
  }
#pragma unroll
  for (int nj = 0; nj < 4; ++nj) {
    int col = w * 64 + nj * 16 + (lane & 15);
    float bias = eb1[col];
#pragma unroll
    for (int mi = 0; mi < 4; ++mi)
#pragma unroll
      for (int r = 0; r < 8; ++r) {
        int row = mi * 16 + r + hi8;
        sH[row * PH + col] = (bf16)tanhf(acc[mi * 4 + nj][r] + bias);
      }
  }
  __syncthreads();

  // ------------- stage 2: mu_z / lv_z = h @ enc_w{mu,lv};  z, KL ----------
  v8f amu[2], alv[2];
  amu[0] = amu[1] = alv[0] = alv[1] = v8f_zero();
  stageWT(ewmu, 0, 0, DZ, sMu0, 32, DZ, PW, tid);
  stageWT(ewlv, 0, 0, DZ, sLv0, 32, DZ, PW, tid);
  __syncthreads();
  {
    const int nbA = ((w & 1) * 2 + 0) * 16;
    const int nbB = ((w & 1) * 2 + 1) * 16;
    for (int c = 0; c < DH / 32; ++c) {
      const bf16* mc = (c & 1) ? sMu1 : sMu0;
      const bf16* lc = (c & 1) ? sLv1 : sLv0;
      if (c + 1 < DH / 32) {
        stageWT(ewmu, (c + 1) * 32, 0, DZ, (c & 1) ? sMu0 : sMu1, 32, DZ, PW, tid);
        stageWT(ewlv, (c + 1) * 32, 0, DZ, (c & 1) ? sLv0 : sLv1, 32, DZ, PW, tid);
      }
      v16bf a   = ldsA(sH, (w >> 1) * 16 + (lane & 15), c * 32, lane, PH);
      v16bf bm0 = ldsB(mc, nbA, 0, lane, PW);
      v16bf bl0 = ldsB(lc, nbA, 0, lane, PW);
      v16bf bm1 = ldsB(mc, nbB, 0, lane, PW);
      v16bf bl1 = ldsB(lc, nbB, 0, lane, PW);
      amu[0] = wmma_bf16(a, bm0, amu[0]);
      alv[0] = wmma_bf16(a, bl0, alv[0]);
      amu[1] = wmma_bf16(a, bm1, amu[1]);
      alv[1] = wmma_bf16(a, bl1, alv[1]);
      __syncthreads();
    }
  }
#pragma unroll
  for (int t = 0; t < 2; ++t) {
    int col = ((w & 1) * 2 + t) * 16 + (lane & 15);
    float bm = ebmu[col], bl = eblv[col];
#pragma unroll
    for (int r = 0; r < 8; ++r) {
      int row  = (w >> 1) * 16 + r + hi8;
      float mu = amu[t][r] + bm;
      float lv = alv[t][r] + bl;
      float zv = mu + expf(0.5f * lv) * eps[(size_t)(row0 + row) * DZ + col];
      sZ[row * PZ + col] = (bf16)zv;
      atomicAdd(&sKL[row], 1.0f + lv - mu * mu - expf(lv));   // ds_add_f32
    }
  }
  __syncthreads();

  // ---------------- stage 3: hd = tanh(z @ dec_w1 + b1) -------------------
  stageWT(dw1, 0, 0, DH, sW0, DZ, DH, PW3, tid);   // 64-k slice spans W region
  __syncthreads();
#pragma unroll
  for (int t = 0; t < 16; ++t) acc[t] = v8f_zero();
  gemm_4x4(sZ, PZ, sW0, PW3, 0, 0, acc, w, lane);
  gemm_4x4(sZ, PZ, sW0, PW3, 32, 32, acc, w, lane);
#pragma unroll
  for (int nj = 0; nj < 4; ++nj) {
    int col = w * 64 + nj * 16 + (lane & 15);
    float bias = db1[col];
#pragma unroll
    for (int mi = 0; mi < 4; ++mi)
#pragma unroll
      for (int r = 0; r < 8; ++r) {
        int row = mi * 16 + r + hi8;
        sH[row * PH + col] = (bf16)tanhf(acc[mi * 4 + nj][r] + bias);  // h->hd
      }
  }
  __syncthreads();

  // -------- stage 4a: logits_mu = hd @ dec_wmu + b  (two 512-col halves) --
  for (int h2 = 0; h2 < 2; ++h2) {
#pragma unroll
    for (int t = 0; t < 16; ++t) acc[t] = v8f_zero();
    stageWT(dwmu, 0, h2 * 512, DIN, sW0, 32, 512, PW, tid);
    __syncthreads();
    for (int c = 0; c < DH / 32; ++c) {
      const bf16* Wc = (c & 1) ? sW1 : sW0;
      if (c + 1 < DH / 32)
        stageWT(dwmu, (c + 1) * 32, h2 * 512, DIN, (c & 1) ? sW0 : sW1,
                32, 512, PW, tid);
      gemm_4x4(sH, PH, Wc, PW, c * 32, 0, acc, w, lane);
      __syncthreads();
    }
#pragma unroll
    for (int nj = 0; nj < 4; ++nj) {
      int col = h2 * 512 + w * 64 + nj * 16 + (lane & 15);
      float bias = dbmu[col];
#pragma unroll
      for (int mi = 0; mi < 4; ++mi)
#pragma unroll
        for (int r = 0; r < 8; ++r) {
          int row = mi * 16 + r + hi8;
          sLog[row * PL + col] = (bf16)(acc[mi * 4 + nj][r] + bias);
        }
    }
    __syncthreads();
  }

  // ---------------- softmax row stats (max, sum of exp) -------------------
  {
    int rr = tid >> 2, part = tid & 3;
    const v8bf* pv = (const v8bf*)(sLog + rr * PL + part * 256);  // 16B aligned
    float m = -3.0e38f;
    for (int cc = 0; cc < 32; ++cc) {
      v8bf vv = pv[cc];
#pragma unroll
      for (int j = 0; j < 8; ++j) m = fmaxf(m, (float)vv[j]);
    }
    sRed[tid] = m;
    __syncthreads();
    if (tid < TM)
      sMax[tid] = fmaxf(fmaxf(sRed[tid * 4], sRed[tid * 4 + 1]),
                        fmaxf(sRed[tid * 4 + 2], sRed[tid * 4 + 3]));
    __syncthreads();
    float mm = sMax[rr], s = 0.f;
    for (int cc = 0; cc < 32; ++cc) {
      v8bf vv = pv[cc];
#pragma unroll
      for (int j = 0; j < 8; ++j) s += expf((float)vv[j] - mm);
    }
    sRed[tid] = s;
    __syncthreads();
    if (tid < TM)
      sSum[tid] = sRed[tid * 4] + sRed[tid * 4 + 1] +
                  sRed[tid * 4 + 2] + sRed[tid * 4 + 3];
    __syncthreads();
  }

  // -------- stage 4b: lv_x = hd @ dec_wlv + b; fuse log-likelihood --------
  float rs[4][8];
#pragma unroll
  for (int mi = 0; mi < 4; ++mi)
#pragma unroll
    for (int r = 0; r < 8; ++r) rs[mi][r] = 0.f;
  for (int h2 = 0; h2 < 2; ++h2) {
#pragma unroll
    for (int t = 0; t < 16; ++t) acc[t] = v8f_zero();
    stageWT(dwlv, 0, h2 * 512, DIN, sW0, 32, 512, PW, tid);
    __syncthreads();
    for (int c = 0; c < DH / 32; ++c) {
      const bf16* Wc = (c & 1) ? sW1 : sW0;
      if (c + 1 < DH / 32)
        stageWT(dwlv, (c + 1) * 32, h2 * 512, DIN, (c & 1) ? sW0 : sW1,
                32, 512, PW, tid);
      gemm_4x4(sH, PH, Wc, PW, c * 32, 0, acc, w, lane);
      __syncthreads();
    }
#pragma unroll
    for (int nj = 0; nj < 4; ++nj) {
      int col = h2 * 512 + w * 64 + nj * 16 + (lane & 15);
      float bias = dblv[col];
#pragma unroll
      for (int mi = 0; mi < 4; ++mi)
#pragma unroll
        for (int r = 0; r < 8; ++r) {
          int row   = mi * 16 + r + hi8;
          float lv  = acc[mi * 4 + nj][r] + bias;
          float mux = expf((float)sLog[row * PL + col] - sMax[row]) / sSum[row];
          float d   = x[(size_t)(row0 + row) * DIN + col] - mux;
          rs[mi][r] += lv + d * d * expf(-lv) + LOG2PI;
        }
    }
    __syncthreads();
  }
#pragma unroll
  for (int mi = 0; mi < 4; ++mi)
#pragma unroll
    for (int r = 0; r < 8; ++r) {
      int row = mi * 16 + r + hi8;
      atomicAdd(&sLP[row], rs[mi][r]);
    }
  __syncthreads();

  // ------- per-row: elbo term = ln_pxz - kl = -0.5*LP + 0.5*KL ------------
  if (tid < TM) sRed[tid] = -0.5f * sLP[tid] + 0.5f * sKL[tid];
  __syncthreads();
  if (tid == 0) {
    float s = 0.f;
    for (int i = 0; i < TM; ++i) s += sRed[i];
    partial[blockIdx.x] = s;
  }
}

__global__ void vae_reduce(const float* __restrict__ part,
                           float* __restrict__ out, int n, float scale) {
  __shared__ float sm[256];
  float s = 0.f;
  for (int i = threadIdx.x; i < n; i += 256) s += part[i];
  sm[threadIdx.x] = s;
  __syncthreads();
  for (int off = 128; off > 0; off >>= 1) {
    if ((int)threadIdx.x < off) sm[threadIdx.x] += sm[threadIdx.x + off];
    __syncthreads();
  }
  if (threadIdx.x == 0) out[0] = sm[0] * scale;
}

extern "C" void kernel_launch(void* const* d_in, const int* in_sizes, int n_in,
                              void* d_out, int out_size, void* d_ws, size_t ws_size,
                              hipStream_t stream) {
  (void)in_sizes; (void)n_in; (void)out_size; (void)ws_size;
  const float* xn   = (const float*)d_in[0];
  const float* x    = (const float*)d_in[1];
  const float* eps  = (const float*)d_in[2];
  const float* ew1  = (const float*)d_in[3];
  const float* eb1  = (const float*)d_in[4];
  const float* ewmu = (const float*)d_in[5];
  const float* ebmu = (const float*)d_in[6];
  const float* ewlv = (const float*)d_in[7];
  const float* eblv = (const float*)d_in[8];
  const float* dw1  = (const float*)d_in[9];
  const float* db1  = (const float*)d_in[10];
  const float* dwmu = (const float*)d_in[11];
  const float* dbmu = (const float*)d_in[12];
  const float* dwlv = (const float*)d_in[13];
  const float* dblv = (const float*)d_in[14];

  float* partial = (float*)d_ws;           // 1024 floats of scratch
  float* out     = (float*)d_out;
  const int ngrp = Bn / TM;                // 1024 workgroups

  vae_fused<<<ngrp, 256, SMEM_BYTES, stream>>>(xn, x, eps, ew1, eb1, ewmu, ebmu,
                                               ewlv, eblv, dw1, db1, dwmu, dbmu,
                                               dwlv, dblv, partial);
  vae_reduce<<<1, 256, 0, stream>>>(partial, out, ngrp, 1.0f / (float)Bn);
}
